// GNN_LSTM_Model_1340029796416
// MI455X (gfx1250) — matrix operand order repaired and checked by reading, not compiled, not run on hardware
//
#include <hip/hip_runtime.h>

// ---------------- problem constants ----------------
#define NND   256   // nodes per graph
#define FD    128   // node feature dim
#define HD    128   // gnn hidden dim
#define TST   32    // timesteps
#define BAT   16    // batch (sequences)
#define HLD   256   // lstm hidden dim
#define NWG   8     // workgroups cooperating on the LSTM

// padded LDS row strides (in halves) -> 16B aligned rows, bank-conflict padding
#define AS_S  264   // 256 + 8
#define XS_S  136   // 128 + 8
#define WS_S  136
#define BS_S  264

// gfx1250 async global->LDS path (guarded so either toolchain compiles)
#if defined(__has_builtin)
#if __has_builtin(__builtin_amdgcn_global_load_async_to_lds_b128) && \
    __has_builtin(__builtin_amdgcn_s_wait_asynccnt)
#define HAVE_ASYNC_LDS 1
#endif
#endif

typedef __attribute__((ext_vector_type(16))) _Float16 v16h;
typedef __attribute__((ext_vector_type(8)))  _Float16 v8h;
typedef __attribute__((ext_vector_type(8)))  float    v8f;

#ifdef HAVE_ASYNC_LDS
// exact operand types for __builtin_amdgcn_global_load_async_to_lds_b128
typedef int v4i_vs __attribute__((vector_size(16)));
typedef v4i_vs __attribute__((address_space(1)))* async_gptr;   // 64-bit global ptr
typedef v4i_vs __attribute__((address_space(3)))* async_lptr;   // 32-bit LDS ptr
#endif

// ---- WMMA operand loaders (wave32, 16x16x32 f16) ----
__device__ __forceinline__ v16h load_A16(const _Float16* base, int stride,
                                         int mbase, int kk, int lane) {
  const int row = mbase + (lane & 15);
  const int off = kk + ((lane >= 16) ? 16 : 0);
  const v8h lo = *(const v8h*)(base + row * stride + off);
  const v8h hi = *(const v8h*)(base + row * stride + off + 8);
  v16h r;
#pragma unroll
  for (int i = 0; i < 8; ++i) { r[i] = lo[i]; r[i + 8] = hi[i]; }
  return r;
}

__device__ __forceinline__ v16h load_B16(const _Float16* base, int stride,
                                         int nbase, int kk, int lane) {
  const int col = nbase + (lane & 15);
  const int off = kk + ((lane >= 16) ? 16 : 0);
  const v8h lo = *(const v8h*)(base + col * stride + off);
  const v8h hi = *(const v8h*)(base + col * stride + off + 8);
  v16h r;
#pragma unroll
  for (int i = 0; i < 8; ++i) { r[i] = lo[i]; r[i + 8] = hi[i]; }
  return r;
}

__device__ __forceinline__ v16h load_Bg16(const _Float16* __restrict__ W, int K,
                                          int gbase, int kk, int lane) {
  const int g = gbase + (lane & 15);
  const int k = kk + ((lane >= 16) ? 16 : 0);
  const _Float16* p = W + (long)g * K + k;
  const v8h lo = *(const v8h*)(p);
  const v8h hi = *(const v8h*)(p + 8);
  v16h r;
#pragma unroll
  for (int i = 0; i < 8; ++i) { r[i] = lo[i]; r[i + 8] = hi[i]; }
  return r;
}

__device__ __forceinline__ float sigf(float x) { return 1.0f / (1.0f + __expf(-x)); }

// ================= GNN kernel (one workgroup per (b,t) graph) =================
#define GNN_AS_OFF   0
#define GNN_XS_OFF   (GNN_AS_OFF + NND * AS_S)
#define GNN_WS_OFF   (GNN_XS_OFF + NND * XS_S)
#define GNN_BS_OFF   (GNN_WS_OFF + HD  * WS_S)
#define GNN_H_TOTAL  (GNN_BS_OFF + HD  * BS_S)        // 153600 halves
#define GNN_SMEM_BYTES (GNN_H_TOTAL * 2 + (256 + 128 + 128 + 128) * 4)  // 309760

__global__ __launch_bounds__(256)
void gnn_kernel(const int* __restrict__ adj, const float* __restrict__ x,
                const float* __restrict__ W1, const float* __restrict__ b1,
                const float* __restrict__ W2, const float* __restrict__ b2,
                float* __restrict__ emb) {
  extern __shared__ char smem[];
  _Float16* As = (_Float16*)smem + GNN_AS_OFF;   // [256][AS_S]  binary A-hat (f16-exact)
  _Float16* Xs = (_Float16*)smem + GNN_XS_OFF;   // [256][XS_S]  x / h1 (A operand)
  _Float16* Ws = (_Float16*)smem + GNN_WS_OFF;   // [128][WS_S]  W^T (B operand)
  _Float16* Bs = (_Float16*)smem + GNN_BS_OFF;   // [128][BS_S]  dinv-scaled (hW)^T (B operand)
  float* dinv = (float*)((_Float16*)smem + GNN_H_TOTAL);
  float* b1s  = dinv + 256;
  float* b2s  = b1s + 128;
  float* embp = b2s + 128;

  const int tid  = threadIdx.x;
  const int lane = tid & 31;
  const int wave = tid >> 5;
  const int bt   = blockIdx.x;
  const long adj_base = (long)bt * NND * NND;
  const long x_base   = (long)bt * NND * FD;

  // ---- adjacency -> binary f16 A-hat ----
#pragma unroll 4
  for (int i = 0; i < 64; ++i) {
    const int e = (i * 256 + tid) * 4;
    __builtin_prefetch(adj + adj_base + e + 8192, 0, 0);   // global_prefetch_b8
    const int4 a4 = *(const int4*)(adj + adj_base + e);
    const int r = e >> 8, c = e & 255;
    _Float16* d = As + r * AS_S + c;
    d[0] = (_Float16)((a4.x != 0) ? 1.0f : 0.0f);
    d[1] = (_Float16)((a4.y != 0) ? 1.0f : 0.0f);
    d[2] = (_Float16)((a4.z != 0) ? 1.0f : 0.0f);
    d[3] = (_Float16)((a4.w != 0) ? 1.0f : 0.0f);
  }
  // ---- x -> f16 (A operand) ----
#pragma unroll 4
  for (int i = 0; i < 32; ++i) {
    const int e = (i * 256 + tid) * 4;
    const float4 v = *(const float4*)(x + x_base + e);
    const int r = e >> 7, c = e & 127;
    _Float16* d = Xs + r * XS_S + c;
    d[0] = (_Float16)v.x; d[1] = (_Float16)v.y; d[2] = (_Float16)v.z; d[3] = (_Float16)v.w;
  }
  // ---- W1^T -> Ws ----
  for (int i = 0; i < 64; ++i) {
    const int e = i * 256 + tid;
    Ws[(e & 127) * WS_S + (e >> 7)] = (_Float16)W1[e];
  }
  if (tid < 128) { b1s[tid] = b1[tid]; b2s[tid] = b2[tid]; embp[tid] = 0.0f; }
  __syncthreads();

  // ---- self-loop + degree + dinv ----
  {
    As[tid * AS_S + tid] = (_Float16)1.0f;
    float s = 0.0f;
    for (int j = 0; j < NND; ++j) s += (float)As[tid * AS_S + j];
    dinv[tid] = rsqrtf(s);
  }
  __syncthreads();

  // ---- XW = x @ W1 -> dinv-scaled transpose in Bs ----
  for (int tile = wave; tile < 128; tile += 8) {
    const int tm = (tile >> 3) * 16, tn = (tile & 7) * 16;
    v8f acc = {};
#pragma unroll
    for (int kk = 0; kk < FD; kk += 32) {
      const v16h a = load_A16(Xs, XS_S, tm, kk, lane);
      const v16h b = load_B16(Ws, WS_S, tn, kk, lane);
      acc = __builtin_amdgcn_wmma_f32_16x16x32_f16(false, a, false, b, (short)0, acc, false, false);
    }
    const int col = tn + (lane & 15);
    const int mb  = tm + ((lane >= 16) ? 8 : 0);
#pragma unroll
    for (int v = 0; v < 8; ++v) {
      const int m = mb + v;
      Bs[col * BS_S + m] = (_Float16)(dinv[m] * acc[v]);
    }
  }
  __syncthreads();

  // reload Ws with W2^T
  for (int i = 0; i < 64; ++i) {
    const int e = i * 256 + tid;
    Ws[(e & 127) * WS_S + (e >> 7)] = (_Float16)W2[e];
  }

  // ---- h1 = relu(dinv * (A-hat @ Bs) + b1) -> Xs ----
  for (int tile = wave; tile < 128; tile += 8) {
    const int tm = (tile >> 3) * 16, tn = (tile & 7) * 16;
    v8f acc = {};
#pragma unroll
    for (int kk = 0; kk < NND; kk += 32) {
      const v16h a = load_A16(As, AS_S, tm, kk, lane);
      const v16h b = load_B16(Bs, BS_S, tn, kk, lane);
      acc = __builtin_amdgcn_wmma_f32_16x16x32_f16(false, a, false, b, (short)0, acc, false, false);
    }
    const int col = tn + (lane & 15);
    const int mb  = tm + ((lane >= 16) ? 8 : 0);
    const float bb = b1s[col];
#pragma unroll
    for (int v = 0; v < 8; ++v) {
      const int m = mb + v;
      Xs[m * XS_S + col] = (_Float16)fmaxf(dinv[m] * acc[v] + bb, 0.0f);
    }
  }
  __syncthreads();

  // ---- HW = h1 @ W2 -> dinv-scaled transpose in Bs ----
  for (int tile = wave; tile < 128; tile += 8) {
    const int tm = (tile >> 3) * 16, tn = (tile & 7) * 16;
    v8f acc = {};
#pragma unroll
    for (int kk = 0; kk < HD; kk += 32) {
      const v16h a = load_A16(Xs, XS_S, tm, kk, lane);
      const v16h b = load_B16(Ws, WS_S, tn, kk, lane);
      acc = __builtin_amdgcn_wmma_f32_16x16x32_f16(false, a, false, b, (short)0, acc, false, false);
    }
    const int col = tn + (lane & 15);
    const int mb  = tm + ((lane >= 16) ? 8 : 0);
#pragma unroll
    for (int v = 0; v < 8; ++v) {
      const int m = mb + v;
      Bs[col * BS_S + m] = (_Float16)(dinv[m] * acc[v]);
    }
  }
  __syncthreads();

  // ---- h2 + mean pool ----
  for (int tile = wave; tile < 128; tile += 8) {
    const int tm = (tile >> 3) * 16, tn = (tile & 7) * 16;
    v8f acc = {};
#pragma unroll
    for (int kk = 0; kk < NND; kk += 32) {
      const v16h a = load_A16(As, AS_S, tm, kk, lane);
      const v16h b = load_B16(Bs, BS_S, tn, kk, lane);
      acc = __builtin_amdgcn_wmma_f32_16x16x32_f16(false, a, false, b, (short)0, acc, false, false);
    }
    const int col = tn + (lane & 15);
    const int mb  = tm + ((lane >= 16) ? 8 : 0);
    const float bb = b2s[col];
    float s = 0.0f;
#pragma unroll
    for (int v = 0; v < 8; ++v) s += fmaxf(dinv[mb + v] * acc[v] + bb, 0.0f);
    atomicAdd(&embp[col], s);   // ds_add_f32
  }
  __syncthreads();

  if (tid < HD) {
    const int b = bt / TST, t = bt % TST;
    emb[((long)t * BAT + b) * HD + tid] = embp[tid] * (1.0f / 256.0f);
  }
}

// ================= helper kernels =================
__global__ void init_kernel(unsigned* bar, float* outacc) {
  if (threadIdx.x == 0) *bar = 0u;
  if (threadIdx.x < BAT) outacc[threadIdx.x] = 0.0f;
}

__global__ __launch_bounds__(256)
void cvt_f16_kernel(const float* __restrict__ src, _Float16* __restrict__ dst, int n) {
  const int e = (blockIdx.x * 256 + threadIdx.x) * 4;
  if (e < n) {
    const float4 v = *(const float4*)(src + e);
    dst[e + 0] = (_Float16)v.x; dst[e + 1] = (_Float16)v.y;
    dst[e + 2] = (_Float16)v.z; dst[e + 3] = (_Float16)v.w;
  }
}

// ================= LSTM: 8 cooperating workgroups, grid barrier =================
__device__ __forceinline__ void gridbar(unsigned* bar, unsigned& nsync) {
  __syncthreads();
  if (threadIdx.x == 0) {
    __threadfence();                 // release h writes
    nsync += 1;
    atomicAdd(bar, 1u);
    const unsigned target = (unsigned)NWG * nsync;
    while (__hip_atomic_load(bar, __ATOMIC_RELAXED, __HIP_MEMORY_SCOPE_AGENT) < target)
      __builtin_amdgcn_s_sleep(2);
    __threadfence();                 // acquire other WGs' h writes
  }
  __syncthreads();
}

// Gather the full shared hidden state [16][256] f16 from global into a padded
// LDS A-operand buffer [16][AS_S].  When the gfx1250 async builtins are
// available this issues GLOBAL_LOAD_ASYNC_TO_LDS_B128 (ASYNCcnt-tracked,
// no VGPR round-trip); otherwise a plain copy.
__device__ __forceinline__ void gather_h(const _Float16* __restrict__ src,
                                         _Float16* __restrict__ dst) {
  const int tid = threadIdx.x;
#ifdef HAVE_ASYNC_LDS
#pragma unroll
  for (int p = 0; p < 2; ++p) {
    const int c = p * 256 + tid;          // 512 chunks of 16B
    const int b = c >> 5;                 // row (32 chunks per 512B row)
    const int q = (c & 31) * 8;           // half offset within row
    __builtin_amdgcn_global_load_async_to_lds_b128(
        (async_gptr)(unsigned long long)(src + b * HLD + q),
        (async_lptr)(unsigned int)(unsigned long long)(dst + b * AS_S + q),
        0, 0);
  }
  __builtin_amdgcn_s_wait_asynccnt(0);
#else
  for (int i = tid; i < BAT * HLD; i += 256) dst[(i >> 8) * AS_S + (i & 255)] = src[i];
#endif
}

__global__ __launch_bounds__(256)
void lstm_kernel(const float* __restrict__ emb,
                 const _Float16* __restrict__ wih0h, const _Float16* __restrict__ whh0h,
                 const _Float16* __restrict__ wih1h, const _Float16* __restrict__ whh1h,
                 const float* __restrict__ bih0, const float* __restrict__ bhh0,
                 const float* __restrict__ bih1, const float* __restrict__ bhh1,
                 const float* __restrict__ Wfc,  const float* __restrict__ bfc,
                 _Float16* __restrict__ h0g, _Float16* __restrict__ h1g,
                 float* __restrict__ outacc, unsigned* __restrict__ bar,
                 float* __restrict__ out) {
  __shared__ _Float16 h0h[BAT * AS_S];   // full layer-0 hidden, A operand
  __shared__ _Float16 h1h[BAT * AS_S];   // full layer-1 hidden, A operand
  __shared__ _Float16 xth[BAT * XS_S];   // x_t, A operand
  __shared__ float    Gs[BAT * 128];     // this WG's gate slice (4 gates x 32)
  __shared__ float    c0l[BAT * 32], c1l[BAT * 32], h1fl[BAT * 32];
  __shared__ float    b0l[128], b1l[128];

  const int tid  = threadIdx.x;
  const int lane = tid & 31;
  const int wave = tid >> 5;
  const int wg   = blockIdx.x;           // 0..NWG-1
  const int jbase = wg * 32;             // hidden-unit slice owned by this WG
  unsigned nsync = 0;

  if (tid < 128) {
    const int g = tid >> 5, jl = tid & 31;
    const int gi = g * HLD + jbase + jl;
    b0l[tid] = bih0[gi] + bhh0[gi];
    b1l[tid] = bih1[gi] + bhh1[gi];
  }
  for (int i = tid; i < BAT * 32; i += 256) { c0l[i] = 0.0f; c1l[i] = 0.0f; h1fl[i] = 0.0f; }
  for (int i = tid; i < BAT * AS_S; i += 256) { h0h[i] = (_Float16)0.0f; h1h[i] = (_Float16)0.0f; }
  __syncthreads();

  // per-wave gate tile: gate g = wave>>1, half = wave&1
  const int gcol0 = (wave >> 1) * HLD + jbase + (wave & 1) * 16;  // global gate column
  const int lcol  = wave * 16;                                    // local Gs column

  for (int t = 0; t < TST; ++t) {
    // x_t -> LDS (f16); prefetch next step's embeddings
    for (int i = tid; i < BAT * HD; i += 256) {
      xth[(i >> 7) * XS_S + (i & 127)] = (_Float16)emb[((long)t * BAT) * HD + i];
    }
    if (t + 1 < TST) __builtin_prefetch(emb + ((long)(t + 1) * BAT) * HD + tid * 4, 0, 1);
    __syncthreads();

    // ---- layer 0 gates: [x | h0] @ [Wih0 | Whh0]^T (one 16x16 tile per wave) ----
    {
      v8f acc = {};
#pragma unroll
      for (int kk = 0; kk < HD; kk += 32) {
        const v16h a = load_A16(xth, XS_S, 0, kk, lane);
        const v16h b = load_Bg16(wih0h, HD, gcol0, kk, lane);
        acc = __builtin_amdgcn_wmma_f32_16x16x32_f16(false, a, false, b, (short)0, acc, false, false);
      }
#pragma unroll
      for (int kk = 0; kk < HLD; kk += 32) {
        const v16h a = load_A16(h0h, AS_S, 0, kk, lane);
        const v16h b = load_Bg16(whh0h, HLD, gcol0, kk, lane);
        acc = __builtin_amdgcn_wmma_f32_16x16x32_f16(false, a, false, b, (short)0, acc, false, false);
      }
      const int lc = lcol + (lane & 15);
      const int mb = (lane >= 16) ? 8 : 0;
#pragma unroll
      for (int v = 0; v < 8; ++v) Gs[(mb + v) * 128 + lc] = acc[v];
    }
    __syncthreads();

    // ---- layer 0 cell update (slice-local), publish h0 slice ----
    for (int i = tid; i < BAT * 32; i += 256) {
      const int b = i >> 5, jl = i & 31;
      const float iv = sigf (Gs[b * 128 +      jl] + b0l[     jl]);
      const float fv = sigf (Gs[b * 128 + 32 + jl] + b0l[32 + jl]);
      const float gv = tanhf(Gs[b * 128 + 64 + jl] + b0l[64 + jl]);
      const float ov = sigf (Gs[b * 128 + 96 + jl] + b0l[96 + jl]);
      const float cn = fv * c0l[i] + iv * gv;
      const float hn = ov * tanhf(cn);
      c0l[i] = cn;
      h0g[b * HLD + jbase + jl] = (_Float16)hn;
    }
    gridbar(bar, nsync);
    gather_h(h0g, h0h);        // full h0(t) -> LDS (async-to-LDS when available)
    __syncthreads();

    // ---- layer 1 gates: [h0 | h1] @ [Wih1 | Whh1]^T ----
    {
      v8f acc = {};
#pragma unroll
      for (int kk = 0; kk < HLD; kk += 32) {
        const v16h a = load_A16(h0h, AS_S, 0, kk, lane);
        const v16h b = load_Bg16(wih1h, HLD, gcol0, kk, lane);
        acc = __builtin_amdgcn_wmma_f32_16x16x32_f16(false, a, false, b, (short)0, acc, false, false);
      }
#pragma unroll
      for (int kk = 0; kk < HLD; kk += 32) {
        const v16h a = load_A16(h1h, AS_S, 0, kk, lane);
        const v16h b = load_Bg16(whh1h, HLD, gcol0, kk, lane);
        acc = __builtin_amdgcn_wmma_f32_16x16x32_f16(false, a, false, b, (short)0, acc, false, false);
      }
      const int lc = lcol + (lane & 15);
      const int mb = (lane >= 16) ? 8 : 0;
#pragma unroll
      for (int v = 0; v < 8; ++v) Gs[(mb + v) * 128 + lc] = acc[v];
    }
    __syncthreads();

    // ---- layer 1 cell update, publish h1 slice ----
    for (int i = tid; i < BAT * 32; i += 256) {
      const int b = i >> 5, jl = i & 31;
      const float iv = sigf (Gs[b * 128 +      jl] + b1l[     jl]);
      const float fv = sigf (Gs[b * 128 + 32 + jl] + b1l[32 + jl]);
      const float gv = tanhf(Gs[b * 128 + 64 + jl] + b1l[64 + jl]);
      const float ov = sigf (Gs[b * 128 + 96 + jl] + b1l[96 + jl]);
      const float cn = fv * c1l[i] + iv * gv;
      const float hn = ov * tanhf(cn);
      c1l[i] = cn;
      h1fl[i] = hn;                                   // f32 kept for final fc
      h1g[b * HLD + jbase + jl] = (_Float16)hn;
    }
    gridbar(bar, nsync);
    gather_h(h1g, h1h);        // full h1(t) -> LDS (recurrent input for next step)
    __syncthreads();
  }

  // ---- fc + sigmoid: partial dot over this WG's slice ----
  if (tid < BAT) {
    float s = 0.0f;
    for (int jl = 0; jl < 32; ++jl) s += h1fl[tid * 32 + jl] * Wfc[jbase + jl];
    atomicAdd(&outacc[tid], s);
  }
  gridbar(bar, nsync);
  if (wg == 0 && tid < BAT) out[tid] = sigf(outacc[tid] + bfc[0]);
}

// ================= host glue =================
// workspace layout (bytes)
#define WS_EMB    0                       // [T][B][H] f32      (262144)
#define WS_H0G    262144                  // [B][HL]   f16      (8192)
#define WS_H1G    270336                  // [B][HL]   f16      (8192)
#define WS_OUTACC 278528                  // [B]       f32      (64)
#define WS_BAR    278592                  // counter            (64)
#define WS_WIH0H  278656                  // [1024][128] f16    (262144)
#define WS_WHH0H  540800                  // [1024][256] f16    (524288)
#define WS_WIH1H  1065088                 // [1024][256] f16    (524288)
#define WS_WHH1H  1589376                 // [1024][256] f16    (524288)  end=2113664

extern "C" void kernel_launch(void* const* d_in, const int* in_sizes, int n_in,
                              void* d_out, int out_size, void* d_ws, size_t ws_size,
                              hipStream_t stream) {
  const int*   adj  = (const int*)  d_in[0];
  const float* x    = (const float*)d_in[1];
  const float* W1   = (const float*)d_in[2];
  const float* b1   = (const float*)d_in[3];
  const float* W2   = (const float*)d_in[4];
  const float* b2   = (const float*)d_in[5];
  const float* Wih0 = (const float*)d_in[6];
  const float* Whh0 = (const float*)d_in[7];
  const float* bih0 = (const float*)d_in[8];
  const float* bhh0 = (const float*)d_in[9];
  const float* Wih1 = (const float*)d_in[10];
  const float* Whh1 = (const float*)d_in[11];
  const float* bih1 = (const float*)d_in[12];
  const float* bhh1 = (const float*)d_in[13];
  const float* Wfc  = (const float*)d_in[14];
  const float* bfc  = (const float*)d_in[15];

  char* ws = (char*)d_ws;
  float*     emb    = (float*)    (ws + WS_EMB);
  _Float16*  h0g    = (_Float16*) (ws + WS_H0G);
  _Float16*  h1g    = (_Float16*) (ws + WS_H1G);
  float*     outacc = (float*)    (ws + WS_OUTACC);
  unsigned*  bar    = (unsigned*) (ws + WS_BAR);
  _Float16*  wih0h  = (_Float16*) (ws + WS_WIH0H);
  _Float16*  whh0h  = (_Float16*) (ws + WS_WHH0H);
  _Float16*  wih1h  = (_Float16*) (ws + WS_WIH1H);
  _Float16*  whh1h  = (_Float16*) (ws + WS_WHH1H);
  float*     outp   = (float*)d_out;

  init_kernel<<<1, 32, 0, stream>>>(bar, outacc);
  cvt_f16_kernel<<<128, 256, 0, stream>>>(Wih0, wih0h, 1024 * 128);
  cvt_f16_kernel<<<256, 256, 0, stream>>>(Whh0, whh0h, 1024 * 256);
  cvt_f16_kernel<<<256, 256, 0, stream>>>(Wih1, wih1h, 1024 * 256);
  cvt_f16_kernel<<<256, 256, 0, stream>>>(Whh1, whh1h, 1024 * 256);
  gnn_kernel<<<BAT * TST, 256, GNN_SMEM_BYTES, stream>>>(adj, x, W1, b1, W2, b2, emb);
  lstm_kernel<<<NWG, 256, 0, stream>>>(emb, wih0h, whh0h, wih1h, whh1h,
                                       bih0, bhh0, bih1, bhh1, Wfc, bfc,
                                       h0g, h1g, outacc, bar, outp);
}
